// KANLinear2D_47742856462889
// MI455X (gfx1250) — matrix operand order repaired and checked by reading, not compiled, not run on hardware
//
#include <hip/hip_runtime.h>

// ---------------------------------------------------------------------------
// KANLinear2D fused as one bf16 WMMA GEMM:
//   out(16384x512) = [silu(x) | spline(x)] (16384x1024, bf16)
//                    @ [base_weight | spline_weight]^T (1024x512, bf16)
// f32 accumulation via v_wmma_f32_16x16x32_bf16 (CDNA5 / gfx1250, wave32).
// Global->LDS staging via global_load_async_to_lds_b128 (ASYNCcnt-tracked).
// ---------------------------------------------------------------------------

typedef __bf16 bf16_t;
typedef __attribute__((ext_vector_type(16))) __bf16 v16bf;
typedef __attribute__((ext_vector_type(8)))  __bf16 v8bf;
typedef __attribute__((ext_vector_type(8)))  float  v8f;
typedef __attribute__((ext_vector_type(4)))  int    v4i;

#define M_TOTAL 16384   // 256*64
#define N_OUT   512
#define K_IN    512
#define K_TOT   1024    // [silu | spline]

#define TILE_M 128
#define TILE_N 128
#define TILE_K 32
#define LDSS   40       // LDS row stride in bf16 elems (32 data + 8 pad = 80B, 16B-aligned)

#if defined(__has_builtin)
#if __has_builtin(__builtin_amdgcn_global_load_async_to_lds_b128)
#define HAS_ASYNC_LDS 1
#endif
#endif

__device__ __forceinline__ void copy16_g2l(const bf16_t* g, bf16_t* l) {
#ifdef HAS_ASYNC_LDS
  // async DMA: global -> LDS, tracked by ASYNCcnt.
  // Builtin signature (from hipcc diagnostic): (v4i __device__*, v4i LDS*, Ii, Ii)
  __builtin_amdgcn_global_load_async_to_lds_b128(
      (__attribute__((address_space(1))) v4i*)g,
      (__attribute__((address_space(3))) v4i*)l, 0, 0);
#else
  *(uint4*)l = *(const uint4*)g;   // global_load_b128 + ds_store_b128
#endif
}

__device__ __forceinline__ void wait_async_all() {
#ifdef HAS_ASYNC_LDS
#if __has_builtin(__builtin_amdgcn_s_wait_asynccnt)
  __builtin_amdgcn_s_wait_asynccnt(0);
#else
  asm volatile("s_wait_asynccnt 0" ::: "memory");
#endif
#endif
}

union Frag16 { v16bf v; v8bf h[2]; };

// ---------------------------------------------------------------------------
// Stage 1: activations. A[m, 0:512] = bf16(silu(x)); A[m, 512:1024] = bf16(spl)
// Cubic B-spline on the uniform extended grid g(j) = -1 + (j-3)*0.4, j=0..11.
// ---------------------------------------------------------------------------
__global__ __launch_bounds__(256)
void kan_prep_kernel(const float* __restrict__ x,
                     const float* __restrict__ bsw,   // (512, 8)
                     bf16_t* __restrict__ A) {
  const int idx = (int)blockIdx.x * 256 + (int)threadIdx.x;   // < M_TOTAL*K_IN
  const int m = idx >> 9;
  const int i = idx & (K_IN - 1);
  const float v = x[idx];

  // base path: silu
  const float sig = 1.0f / (1.0f + __expf(-v));
  A[(size_t)m * K_TOT + i] = (bf16_t)(v * sig);

  // spline path: order-0 indicators then 3 recursion steps (all grid consts fold)
  const float h = 0.4f;
  float b[11];
#pragma unroll
  for (int j = 0; j < 11; ++j) {
    const float g0 = -1.0f + (float)(j - 3) * h;
    const float g1 = g0 + h;
    b[j] = (v >= g0 && v < g1) ? 1.0f : 0.0f;
  }
#pragma unroll
  for (int k = 1; k <= 3; ++k) {
    const float inv = 1.0f / ((float)k * h);
#pragma unroll
    for (int j = 0; j < 11 - k; ++j) {
      const float gj = -1.0f + (float)(j - 3) * h;          // grid[j]
      const float gk = -1.0f + (float)(j + k - 2) * h;      // grid[j+k+1]
      b[j] = (v - gj) * inv * b[j] + (gk - v) * inv * b[j + 1];
    }
  }
  const float* w = bsw + i * 8;
  float spl = 0.0f;
#pragma unroll
  for (int k = 0; k < 8; ++k) spl = __builtin_fmaf(w[k], b[k], spl);
  A[(size_t)m * K_TOT + K_IN + i] = (bf16_t)spl;
}

// ---------------------------------------------------------------------------
// Stage 2: pack Wc(n, 0:512)=base_weight(n,:), Wc(n, 512:1024)=spline_weight(n,:)
// ---------------------------------------------------------------------------
__global__ __launch_bounds__(256)
void kan_packw_kernel(const float* __restrict__ bw,
                      const float* __restrict__ sw,
                      bf16_t* __restrict__ Wc) {
  const int idx = (int)blockIdx.x * 256 + (int)threadIdx.x;   // < N_OUT*K_TOT
  const int n = idx >> 10;
  const int k = idx & (K_TOT - 1);
  const float v = (k < K_IN) ? bw[n * K_IN + k] : sw[n * K_IN + (k - K_IN)];
  Wc[idx] = (bf16_t)v;
}

// ---------------------------------------------------------------------------
// Stage 3: C(M,N) = A(M,K) * Wc(N,K)^T, bf16 in / f32 out.
// WG = 256 thr = 8 waves; block tile 128x128; wave tile 32x64 = 2x4 WMMA accs.
// ---------------------------------------------------------------------------
__global__ __launch_bounds__(256)
void kan_gemm_kernel(const bf16_t* __restrict__ A,
                     const bf16_t* __restrict__ W,
                     float* __restrict__ C) {
  __shared__ alignas(16) bf16_t sA[2][TILE_M * LDSS];
  __shared__ alignas(16) bf16_t sB[2][TILE_N * LDSS];

  const int tid    = (int)threadIdx.x;
  const int lane   = tid & 31;
  const int wave   = tid >> 5;
  const int waveM  = wave & 3;              // 4 waves along M -> 128
  const int waveN  = wave >> 2;             // 2 waves along N -> 128
  const int lrow   = lane & 15;
  const int hsel   = lane >> 4;             // which half-K group this lane owns
  const int blockM = (int)blockIdx.x * TILE_M;
  const int blockN = (int)blockIdx.y * TILE_N;

  v8f acc[2][4];
#pragma unroll
  for (int mi = 0; mi < 2; ++mi)
#pragma unroll
    for (int ni = 0; ni < 4; ++ni)
      acc[mi][ni] = (v8f){0.f, 0.f, 0.f, 0.f, 0.f, 0.f, 0.f, 0.f};

  // tile loader: 128 rows x 64B each for A and B -> 512 16B chunks apiece,
  // 2 chunks per thread per matrix.
  auto load_tiles = [&](int kt, int buf) {
#pragma unroll
    for (int r = 0; r < 2; ++r) {
      const int c   = tid + r * 256;        // 0..511
      const int row = c >> 2;               // 0..127
      const int col = (c & 3) * 8;          // bf16 elems within the K-tile
      copy16_g2l(A + (size_t)(blockM + row) * K_TOT + kt * TILE_K + col,
                 &sA[buf][row * LDSS + col]);
      copy16_g2l(W + (size_t)(blockN + row) * K_TOT + kt * TILE_K + col,
                 &sB[buf][row * LDSS + col]);
    }
  };

  load_tiles(0, 0);

  const int KSTEPS = K_TOT / TILE_K;        // 32
  for (int kt = 0; kt < KSTEPS; ++kt) {
    const int buf = kt & 1;
    wait_async_all();
    __syncthreads();
    if (kt + 1 < KSTEPS) load_tiles(kt + 1, buf ^ 1);   // overlap with WMMA below

    // A fragment (16x32 bf16): lanes 0-15 row M=l, K={0..7,16..23};
    // lanes 16-31 row M=l-16, K={8..15,24..31}  (ISA 7.12.2)
    Frag16 af[2];
#pragma unroll
    for (int mi = 0; mi < 2; ++mi) {
      const bf16_t* p = &sA[buf][(waveM * 32 + mi * 16 + lrow) * LDSS];
      af[mi].h[0] = *(const v8bf*)(p + hsel * 8);        // K = 8*hsel ..
      af[mi].h[1] = *(const v8bf*)(p + 16 + hsel * 8);   // K = 16 + 8*hsel ..
    }
    // B fragment (32x16 bf16): lanes 0-15 hold K={0..7,16..23}, lanes 16-31 K={8..15,24..31},
    // N = lane&15 (same K order as A fragment)
    Frag16 bfm[4];
#pragma unroll
    for (int ni = 0; ni < 4; ++ni) {
      const bf16_t* p = &sB[buf][(waveN * 64 + ni * 16 + lrow) * LDSS];
      bfm[ni].h[0] = *(const v8bf*)(p + hsel * 8);
      bfm[ni].h[1] = *(const v8bf*)(p + 16 + hsel * 8);
    }

#pragma unroll
    for (int mi = 0; mi < 2; ++mi)
#pragma unroll
      for (int ni = 0; ni < 4; ++ni)
        acc[mi][ni] = __builtin_amdgcn_wmma_f32_16x16x32_bf16(
            false, af[mi].v, false, bfm[ni].v,
            (short)0, acc[mi][ni], false, false);

    __syncthreads();
  }

  // C/D layout: VGPR r -> (M = r + 8*hsel, N = lane&15) within each 16x16 tile
#pragma unroll
  for (int mi = 0; mi < 2; ++mi) {
#pragma unroll
    for (int ni = 0; ni < 4; ++ni) {
      const int mb = blockM + waveM * 32 + mi * 16 + hsel * 8;
      const int nn = blockN + waveN * 64 + ni * 16 + lrow;
#pragma unroll
      for (int r = 0; r < 8; ++r)
        C[(size_t)(mb + r) * N_OUT + nn] = acc[mi][ni][r];
    }
  }
}

// ---------------------------------------------------------------------------
extern "C" void kernel_launch(void* const* d_in, const int* in_sizes, int n_in,
                              void* d_out, int out_size, void* d_ws, size_t ws_size,
                              hipStream_t stream) {
  const float* x   = (const float*)d_in[0];   // (256,64,512)
  const float* bw  = (const float*)d_in[1];   // base_weight   (512,512)
  const float* sw  = (const float*)d_in[2];   // spline_weight (512,512)
  const float* bsw = (const float*)d_in[3];   // B_spline_weight (512,8)
  float* out = (float*)d_out;                 // (256,64,512) f32

  // workspace: A (16384x1024 bf16) then Wc (512x1024 bf16) ~= 34.6 MB
  bf16_t* A  = (bf16_t*)d_ws;
  bf16_t* Wc = A + (size_t)M_TOTAL * K_TOT;

  kan_prep_kernel<<<(M_TOTAL * K_IN) / 256, 256, 0, stream>>>(x, bsw, A);
  kan_packw_kernel<<<(N_OUT * K_TOT) / 256, 256, 0, stream>>>(bw, sw, Wc);

  dim3 grid(M_TOTAL / TILE_M, N_OUT / TILE_N);   // (128, 4)
  kan_gemm_kernel<<<grid, 256, 0, stream>>>(A, Wc, out);
}
